// SelfAttention_68083821576563
// MI455X (gfx1250) — compile-verified, matrix operand order checked by reading, not run on hardware
//
#include <hip/hip_runtime.h>
#include <hip/hip_bf16.h>
#include <math.h>

// ---------------------------------------------------------------------------
// Self-attention (B=4, T=2048, D=2048, H=16, hd=128), causal.
// All matmuls via v_wmma_f32_16x16x32_f16 (fp32 accumulate).
// Pipeline: qkv GEMM (V stored transposed) -> RoPE (in place) ->
//           flash attention -> out GEMM.
// Workspace layout (f16): q[16M] k[16M] vT[16M] y[16M]  = 128 MB total.
//   q,k : [B,H,T,hd]      vT : [B,H,hd,T]      y : [B,T,D]
// ---------------------------------------------------------------------------

typedef __attribute__((ext_vector_type(16))) _Float16 v16h;
typedef __attribute__((ext_vector_type(8)))  float    v8f;

#define TSEQ 2048
#define DMODEL 2048
#define NHEAD 16
#define HD 128
#define BATCH 4

static __device__ __forceinline__ v8f wmma_f16(v16h a, v16h b, v8f c) {
    // D = A(16x32 f16) * B(32x16 f16) + C(16x16 f32)
    return __builtin_amdgcn_wmma_f32_16x16x32_f16(
        false, a, false, b, (short)0, c, false, false);
}

// K-offset pattern shared by 16-bit A and B fragment layouts (ISA 7.12.2):
//   vgpr v, half h -> k = (v&3)*2 + (v>=4 ? 16 : 0) + 8*(lane>=16) + h
static __device__ __forceinline__ int kpat(int v) {
    return ((v & 3) * 2) + ((v >= 4) ? 16 : 0);
}

// A fragment (16x32) from row-major fp32 matrix, convert to f16.
static __device__ __forceinline__ v16h load_a_f32(const float* A, int lda,
                                                  int row0, int k0) {
    int lane = threadIdx.x & 31;
    int m = lane & 15, g = lane >> 4;
    const float* p = A + (size_t)(row0 + m) * lda + k0 + 8 * g;
    v16h out;
#pragma unroll
    for (int v = 0; v < 8; ++v) {
        int kb = kpat(v);
        out[2 * v]     = (_Float16)p[kb];
        out[2 * v + 1] = (_Float16)p[kb + 1];
    }
    return out;
}

// A fragment (16x32) from row-major f16 matrix.
static __device__ __forceinline__ v16h load_a_f16(const _Float16* A, int lda,
                                                  int row0, int k0) {
    int lane = threadIdx.x & 31;
    int m = lane & 15, g = lane >> 4;
    const _Float16* p = A + (size_t)(row0 + m) * lda + k0 + 8 * g;
    v16h out;
#pragma unroll
    for (int v = 0; v < 8; ++v) {
        int kb = kpat(v);
        out[2 * v]     = p[kb];
        out[2 * v + 1] = p[kb + 1];
    }
    return out;
}

// B fragment (32x16) where B[k][n] = W[n0+n][k0+k]  (W row-major fp32; B = W^T).
static __device__ __forceinline__ v16h load_bt_f32(const float* W, int ldw,
                                                   int n0, int k0) {
    int lane = threadIdx.x & 31;
    int n = lane & 15, g = lane >> 4;
    const float* p = W + (size_t)(n0 + n) * ldw + k0 + 8 * g;
    v16h out;
#pragma unroll
    for (int v = 0; v < 8; ++v) {
        int kb = kpat(v);
        out[2 * v]     = (_Float16)p[kb];
        out[2 * v + 1] = (_Float16)p[kb + 1];
    }
    return out;
}

// B fragment (32x16) where B[k][n] = Km[n0+n][k0+k]  (Km row-major f16; B = Km^T).
// Per-lane reads are two contiguous 8xf16 groups -> vectorizes to b128 loads.
static __device__ __forceinline__ v16h load_bt_f16(const _Float16* Km, int ldk,
                                                   int n0, int k0) {
    int lane = threadIdx.x & 31;
    int n = lane & 15, g = lane >> 4;
    const _Float16* p = Km + (size_t)(n0 + n) * ldk + k0 + 8 * g;
    v16h out;
#pragma unroll
    for (int v = 0; v < 8; ++v) {
        int kb = kpat(v);
        out[2 * v]     = p[kb];
        out[2 * v + 1] = p[kb + 1];
    }
    return out;
}

// ---------------------------------------------------------------------------
// Kernel 1: qkv = x @ w_qkv^T; scatter f16 results into q/k ([B,H,T,hd]) and
// V transposed ([B,H,hd,T]) so the attention PV fragments load contiguously.
// Block = 256 threads (8 waves, 2x4), block tile 128(M) x 128(N).
// ---------------------------------------------------------------------------
__global__ __launch_bounds__(256) void qkv_gemm_kernel(
    const float* __restrict__ x, const float* __restrict__ wqkv,
    _Float16* __restrict__ qb, _Float16* __restrict__ kb,
    _Float16* __restrict__ vtb) {
    int m0 = blockIdx.x * 128, n0 = blockIdx.y * 128;
    int wave = threadIdx.x >> 5;
    int wm0 = m0 + (wave & 1) * 64;
    int wn0 = n0 + (wave >> 1) * 32;

    v8f acc[4][2] = {};
#pragma unroll 1
    for (int k0 = 0; k0 < DMODEL; k0 += 32) {
        v16h a[4], bf[2];
#pragma unroll
        for (int i = 0; i < 4; ++i) a[i] = load_a_f32(x, DMODEL, wm0 + 16 * i, k0);
#pragma unroll
        for (int j = 0; j < 2; ++j) bf[j] = load_bt_f32(wqkv, DMODEL, wn0 + 16 * j, k0);
#pragma unroll
        for (int i = 0; i < 4; ++i)
#pragma unroll
            for (int j = 0; j < 2; ++j) acc[i][j] = wmma_f16(a[i], bf[j], acc[i][j]);
    }

    int lane = threadIdx.x & 31, g = lane >> 4, nn = lane & 15;
#pragma unroll
    for (int i = 0; i < 4; ++i) {
#pragma unroll
        for (int j = 0; j < 2; ++j) {
            int N = wn0 + 16 * j + nn;
            int which = N >> 11;              // 0=q 1=k 2=v (uniform per tile)
            int h = (N >> 7) & 15, d = N & 127;
            if (which == 2) {
                // V transposed: vT[b,h,d,t]
#pragma unroll
                for (int r = 0; r < 8; ++r) {
                    int M = wm0 + 16 * i + r + 8 * g;
                    int b = M >> 11, t = M & (TSEQ - 1);
                    vtb[(((size_t)b * NHEAD + h) * HD + d) * TSEQ + t] =
                        (_Float16)acc[i][j][r];
                }
            } else {
                _Float16* dst = (which == 0) ? qb : kb;
#pragma unroll
                for (int r = 0; r < 8; ++r) {
                    int M = wm0 + 16 * i + r + 8 * g;
                    int b = M >> 11, t = M & (TSEQ - 1);
                    dst[(((size_t)b * NHEAD + h) * TSEQ + t) * HD + d] =
                        (_Float16)acc[i][j][r];
                }
            }
        }
    }
}

// ---------------------------------------------------------------------------
// Kernel 2: RoPE in place on q,k ([B,H,T,hd]); fold 1/sqrt(hd) into q.
// One thread per (b,h,t,d<64) rotation pair.
// ---------------------------------------------------------------------------
__global__ __launch_bounds__(256) void rope_kernel(_Float16* __restrict__ qb,
                                                   _Float16* __restrict__ kb) {
    size_t idx = (size_t)blockIdx.x * blockDim.x + threadIdx.x;
    int d = (int)(idx & 63);
    size_t rest = idx >> 6;
    int t = (int)(rest & (TSEQ - 1));
    size_t bh = rest >> 11;

    float inv = __powf(10000.f, -(float)d * (1.0f / 64.0f));
    float ang = (float)t * inv;
    float c, s;
    __sincosf(ang, &s, &c);  // s = sin, c = cos
    const float scale = 0.08838834764831845f;  // 1/sqrt(128)

    size_t base = (bh * TSEQ + (size_t)t) * HD + d;
    float q1 = (float)qb[base], q2 = (float)qb[base + 64];
    qb[base]      = (_Float16)((q1 * c - q2 * s) * scale);
    qb[base + 64] = (_Float16)((q1 * s + q2 * c) * scale);
    float k1 = (float)kb[base], k2 = (float)kb[base + 64];
    kb[base]      = (_Float16)(k1 * c - k2 * s);
    kb[base + 64] = (_Float16)(k1 * s + k2 * c);
}

// ---------------------------------------------------------------------------
// Kernel 3: causal flash attention. Block = 128 threads (4 waves); each wave
// owns 16 query rows, streams keys in chunks of 32 with online softmax.
// V is consumed from the transposed layout, so both K and V B-fragments are
// contiguous-pair loads (b128-vectorizable).
// ---------------------------------------------------------------------------
__global__ __launch_bounds__(128) void attn_kernel(
    const _Float16* __restrict__ qb, const _Float16* __restrict__ kb,
    const _Float16* __restrict__ vtb, _Float16* __restrict__ yb) {
    int blk = blockIdx.x;
    int qblk = blk & (TSEQ / 64 - 1);
    int bh = blk / (TSEQ / 64);
    int h = bh & (NHEAD - 1), b = bh / NHEAD;
    int wave = threadIdx.x >> 5;
    int q0 = qblk * 64 + wave * 16;

    const _Float16* Q  = qb  + (size_t)bh * TSEQ * HD;
    const _Float16* K  = kb  + (size_t)bh * TSEQ * HD;
    const _Float16* Vt = vtb + (size_t)bh * HD * TSEQ;

    int lane = threadIdx.x & 31, g = lane >> 4, nn = lane & 15;

    // Q fragments for this wave's 16 rows, full hd=128 (4 K-chunks of 32).
    v16h qf[4];
#pragma unroll
    for (int c = 0; c < 4; ++c) qf[c] = load_a_f16(Q, HD, q0, c * 32);

    v8f o[8] = {};
    float mrow[8], lrow[8];
#pragma unroll
    for (int i = 0; i < 8; ++i) { mrow[i] = -1e30f; lrow[i] = 0.f; }

    __shared__ _Float16 pshare[4][16][32];  // per-wave P tile (C->A relayout)
    _Float16* pflat = &pshare[wave][0][0];

    int jmax = q0 + 16;  // causal bound (keys <= q0+15 needed)
#pragma unroll 1
    for (int j0 = 0; j0 < jmax; j0 += 32) {
        // S = Q K^T for 32 keys -> two 16x16 f32 accumulators
        v8f s0 = {}, s1 = {};
#pragma unroll
        for (int c = 0; c < 4; ++c) {
            v16h b0 = load_bt_f16(K, HD, j0, c * 32);
            v16h b1 = load_bt_f16(K, HD, j0 + 16, c * 32);
            s0 = wmma_f16(qf[c], b0, s0);
            s1 = wmma_f16(qf[c], b1, s1);
        }

        // causal mask + online softmax (rows live in vgpr slots, cols in lanes)
        float arow[8];
#pragma unroll
        for (int i = 0; i < 8; ++i) {
            int row = q0 + i + 8 * g;
            float v0 = (j0 + nn <= row) ? s0[i] : -1e30f;
            float v1 = (j0 + 16 + nn <= row) ? s1[i] : -1e30f;
            float mx = fmaxf(v0, v1);
#pragma unroll
            for (int dlt = 1; dlt < 16; dlt <<= 1)
                mx = fmaxf(mx, __shfl_xor(mx, dlt, 32));
            float mnew = fmaxf(mrow[i], mx);
            float alpha = __expf(mrow[i] - mnew);
            float p0 = __expf(v0 - mnew);
            float p1 = __expf(v1 - mnew);
            float ps = p0 + p1;
#pragma unroll
            for (int dlt = 1; dlt < 16; dlt <<= 1)
                ps += __shfl_xor(ps, dlt, 32);
            lrow[i] = lrow[i] * alpha + ps;
            mrow[i] = mnew;
            arow[i] = alpha;
            s0[i] = p0;
            s1[i] = p1;
        }
#pragma unroll
        for (int nt = 0; nt < 8; ++nt)
#pragma unroll
            for (int i = 0; i < 8; ++i) o[nt][i] *= arow[i];

        // C-layout P -> LDS row-major [16][32] -> A-layout fragment.
#pragma unroll
        for (int i = 0; i < 8; ++i) {
            pshare[wave][i + 8 * g][nn]      = (_Float16)s0[i];
            pshare[wave][i + 8 * g][16 + nn] = (_Float16)s1[i];
        }
        v16h pa = load_a_f16(pflat, 32, 0, 0);  // same-wave LDS RAW: hw waits

        // O += P * V   (8 tiles of 16 along hd); B[k=j][n=d] = Vt[d][j]
#pragma unroll
        for (int nt = 0; nt < 8; ++nt) {
            v16h bv = load_bt_f16(Vt, TSEQ, nt * 16, j0);
            o[nt] = wmma_f16(pa, bv, o[nt]);
        }
    }

    // epilogue: normalize and store y as f16 in [B, T, D] layout
    float rl[8];
#pragma unroll
    for (int i = 0; i < 8; ++i) rl[i] = 1.0f / lrow[i];
#pragma unroll
    for (int nt = 0; nt < 8; ++nt) {
#pragma unroll
        for (int i = 0; i < 8; ++i) {
            int t = q0 + i + 8 * g;
            size_t dst = ((size_t)b * TSEQ + t) * DMODEL + h * HD + nt * 16 + nn;
            yb[dst] = (_Float16)(o[nt][i] * rl[i]);
        }
    }
}

// ---------------------------------------------------------------------------
// Kernel 4: out = y @ w_out^T  (y f16, w_out fp32 converted on load, out fp32)
// ---------------------------------------------------------------------------
__global__ __launch_bounds__(256) void out_gemm_kernel(
    const _Float16* __restrict__ y, const float* __restrict__ wout,
    float* __restrict__ out) {
    int m0 = blockIdx.x * 128, n0 = blockIdx.y * 128;
    int wave = threadIdx.x >> 5;
    int wm0 = m0 + (wave & 1) * 64;
    int wn0 = n0 + (wave >> 1) * 32;

    v8f acc[4][2] = {};
#pragma unroll 1
    for (int k0 = 0; k0 < DMODEL; k0 += 32) {
        v16h a[4], bf[2];
#pragma unroll
        for (int i = 0; i < 4; ++i) a[i] = load_a_f16(y, DMODEL, wm0 + 16 * i, k0);
#pragma unroll
        for (int j = 0; j < 2; ++j) bf[j] = load_bt_f32(wout, DMODEL, wn0 + 16 * j, k0);
#pragma unroll
        for (int i = 0; i < 4; ++i)
#pragma unroll
            for (int j = 0; j < 2; ++j) acc[i][j] = wmma_f16(a[i], bf[j], acc[i][j]);
    }

    int lane = threadIdx.x & 31, g = lane >> 4, nn = lane & 15;
#pragma unroll
    for (int i = 0; i < 4; ++i)
#pragma unroll
        for (int j = 0; j < 2; ++j)
#pragma unroll
            for (int r = 0; r < 8; ++r) {
                int M = wm0 + 16 * i + r + 8 * g;
                int N = wn0 + 16 * j + nn;
                out[(size_t)M * DMODEL + N] = acc[i][j][r];
            }
}

// ---------------------------------------------------------------------------
extern "C" void kernel_launch(void* const* d_in, const int* in_sizes, int n_in,
                              void* d_out, int out_size, void* d_ws,
                              size_t ws_size, hipStream_t stream) {
    (void)in_sizes; (void)n_in; (void)out_size; (void)ws_size;
    const float* x    = (const float*)d_in[0];
    // d_in[1] = attn_mask (unused: setup uses is_causal=1 path)
    // d_in[2] = is_causal (always 1 in this problem's setup)
    const float* wqkv = (const float*)d_in[3];
    const float* wout = (const float*)d_in[4];
    float* out = (float*)d_out;

    const size_t NELEM = (size_t)BATCH * NHEAD * TSEQ * HD;  // 16M per buffer
    _Float16* qb   = (_Float16*)d_ws;
    _Float16* kbuf = qb + NELEM;
    _Float16* vtbf = kbuf + NELEM;
    _Float16* ybuf = vtbf + NELEM;

    // 1) QKV projection (WMMA f16)  grid: 8192/128 x 6144/128
    qkv_gemm_kernel<<<dim3(64, 48), 256, 0, stream>>>(x, wqkv, qb, kbuf, vtbf);
    // 2) RoPE on q,k (in place), fold 1/sqrt(hd) into q
    {
        size_t pairs = (size_t)BATCH * NHEAD * TSEQ * 64;
        rope_kernel<<<(unsigned)(pairs / 256), 256, 0, stream>>>(qb, kbuf);
    }
    // 3) causal flash attention  grid: B*H*(T/64)
    attn_kernel<<<BATCH * NHEAD * (TSEQ / 64), 128, 0, stream>>>(qb, kbuf, vtbf,
                                                                 ybuf);
    // 4) output projection  grid: 8192/128 x 2048/128
    out_gemm_kernel<<<dim3(64, 16), 256, 0, stream>>>(ybuf, wout, out);
}
